// QCNNHybrid_65481071403406
// MI455X (gfx1250) — matrix operand order repaired and checked by reading, not compiled, not run on hardware
//
#include <hip/hip_runtime.h>
#include <hip/hip_bf16.h>
#include <math.h>

typedef __attribute__((ext_vector_type(16))) _Float16 v16h;
typedef __attribute__((ext_vector_type(2)))  _Float16 h2;
typedef __attribute__((ext_vector_type(8)))  float    v8f;

#define WAVES_PER_BLOCK 8
#define TPW 8             // iterations per wave; each iteration = 2 x 16-row sub-tiles

__device__ __forceinline__ float fast_exp2(float x) { return __builtin_amdgcn_exp2f(x); }
__device__ __forceinline__ float fast_rcp(float x)  { return __builtin_amdgcn_rcpf(x); }

__device__ __forceinline__ float fast_tanh(float x) {
#if __has_builtin(__builtin_amdgcn_tanhf)
  return __builtin_amdgcn_tanhf(x);
#else
  float e = fast_exp2(x * 2.88539008177793f);
  return 1.0f - 2.0f * fast_rcp(e + 1.0f);
#endif
}
__device__ __forceinline__ float fast_sigmoid(float x) {
  return fast_rcp(1.0f + fast_exp2(x * -1.44269504088896f));
}

// Cross-half (lane i <-> lane i+16) exchange. v_permlanex16_b32 is a pure VALU
// op (no DS pipe / DScnt); identity selectors give the xor-16 swap.
__device__ __forceinline__ int swap_half(int v) {
#if __has_builtin(__builtin_amdgcn_permlanex16)
  return __builtin_amdgcn_permlanex16(v, v, 0x76543210, 0xfedcba98, false, false);
#else
  return __shfl_xor(v, 16, 32);
#endif
}

// ---- Static weight fragment in WMMA *A* layout (transposed compute):
// A = [ W^T | bias ] with shape dout x (din+1), zero-padded to 16 x 32.
// lane m = lane&15 = output-feature row; slot i <-> k = 16*hb + i.
__device__ __forceinline__ v16h make_wA(const float* __restrict__ w,
                                        const float* __restrict__ bias,
                                        int din, int dout, int lane) {
  v16h a;
  int mrow = lane & 15;
  int hb   = (lane >> 4) & 1;
#pragma unroll
  for (int i = 0; i < 16; ++i) {
    int k = (hb << 4) + i;
    float v = 0.0f;
    if (mrow < dout) {
      if (k < din)       v = w[k * dout + mrow];   // W^T[m][k]
      else if (k == din) v = bias[mrow];
    }
    a[i] = (_Float16)v;
  }
  return a;
}

// ---- Build next-layer B fragment (K x N = 32 x 16, K = feature, N = batch col)
// from the D result of the previous layer (C layout: lane n = batch col,
// reg r = feature r + 8*hb), activation already applied in `act`.
// Features 8..DOUT-1 come from the upper half-wave via v_permlanex16.
// Constant 1.0 at K == DOUT activates the bias column of the next weight A.
template<int DOUT>
__device__ __forceinline__ v16h next_b(const v8f& act, int hb) {
  union U { v16h v; h2 d[8]; };
  const h2 one0  = {(_Float16)1.0f, (_Float16)0.0f};
  const h2 zero2 = {(_Float16)0.0f, (_Float16)0.0f};
  constexpr int NP = (DOUT >= 8) ? 4 : (DOUT + 1) / 2;       // own packed dwords
  constexpr int NS = (DOUT > 8) ? (DOUT - 8 + 1) / 2 : 0;    // dwords from other half

  h2 p[4];
#pragma unroll
  for (int j = 0; j < NP; ++j) {
    h2 t; t[0] = (_Float16)act[2 * j]; t[1] = (_Float16)act[2 * j + 1];
    p[j] = t;
  }
  h2 s[NS > 0 ? NS : 1];
#pragma unroll
  for (int j = 0; j < NS; ++j) {
    int pw = __builtin_bit_cast(int, p[j]);
    s[j] = __builtin_bit_cast(h2, swap_half(pw));             // half-wave swap (VALU)
  }

  U ub;
#pragma unroll
  for (int j = 0; j < 8; ++j) {
    h2 lo = zero2;                       // value for lanes 0-15 (K = 0..15)
    if (j < 4) {
      if (j < NP) lo = p[j];
      if (DOUT < 8 && j == DOUT / 2) lo = one0;
    } else {
      if (j - 4 < NS) lo = s[j - 4];
      if (DOUT >= 8 && DOUT < 16 && j == DOUT / 2) lo = one0;
    }
    h2 hi = (DOUT == 16 && j == 0) ? one0 : zero2;   // lanes 16-31 (K = 16..31)
    ub.d[j] = hb ? hi : lo;
  }
  return ub.v;
}

// =======================  workspace init  =======================
__global__ void zero_ws(float* ws) {
  if (threadIdx.x < 16) ws[threadIdx.x] = 0.0f;
}

// =======================  batch statistics  =======================
__global__ void __launch_bounds__(256) bn_stats(const float* __restrict__ x,
                                                float* __restrict__ ws, int nrows) {
  __shared__ float acc[16];
  float s[8], q[8];
#pragma unroll
  for (int k = 0; k < 8; ++k) { s[k] = 0.0f; q[k] = 0.0f; }

  int stride = gridDim.x * blockDim.x;
  for (int row = blockIdx.x * blockDim.x + threadIdx.x; row < nrows; row += stride) {
    const float4* p = (const float4*)(x + (size_t)row * 8);
    float4 lo = p[0], hi = p[1];
    float v[8] = {lo.x, lo.y, lo.z, lo.w, hi.x, hi.y, hi.z, hi.w};
#pragma unroll
    for (int k = 0; k < 8; ++k) { s[k] += v[k]; q[k] += v[k] * v[k]; }
  }
#pragma unroll
  for (int off = 16; off >= 1; off >>= 1) {
#pragma unroll
    for (int k = 0; k < 8; ++k) {
      s[k] += __shfl_xor(s[k], off, 32);
      q[k] += __shfl_xor(q[k], off, 32);
    }
  }
  if (threadIdx.x < 16) acc[threadIdx.x] = 0.0f;
  __syncthreads();
  if ((threadIdx.x & 31) == 0) {
#pragma unroll
    for (int k = 0; k < 8; ++k) {
      atomicAdd(&acc[k],     s[k]);
      atomicAdd(&acc[8 + k], q[k]);
    }
  }
  __syncthreads();
  if (threadIdx.x < 16) atomicAdd(&ws[threadIdx.x], acc[threadIdx.x]);
}

// =======================  fused MLP chain (transposed) =======================
// H_next^T = [W^T|b] x [H^T;1]: weights are the static A operand, activations
// are the B operand; inter-layer layout conversion is a VALU half-wave swap.
// No LDS in this kernel.  Two sub-tiles in flight fill WMMA hazard slots.
__global__ void __launch_bounds__(256, 1) qcnn_fused(
    const float* __restrict__ x,
    const float* __restrict__ bn_g, const float* __restrict__ bn_b,
    const float* __restrict__ w_fm, const float* __restrict__ b_fm,
    const float* __restrict__ w_c1, const float* __restrict__ b_c1,
    const float* __restrict__ w_p1, const float* __restrict__ b_p1,
    const float* __restrict__ w_c2, const float* __restrict__ b_c2,
    const float* __restrict__ w_p2, const float* __restrict__ b_p2,
    const float* __restrict__ w_c3, const float* __restrict__ b_c3,
    const float* __restrict__ w_r,  const float* __restrict__ b_r,
    const float* __restrict__ w_h,  const float* __restrict__ b_h,
    const float* __restrict__ stats, float* __restrict__ out, int nrows)
{
  const int lane = threadIdx.x & 31;
  const int wave = threadIdx.x >> 5;
  const int n    = lane & 15;       // batch column within sub-tile
  const int hb   = lane >> 4;

  // ---- BatchNorm affine constants per feature ----
  float scale[8], shift[8];
  const float invB = 1.0f / (float)nrows;
#pragma unroll
  for (int k = 0; k < 8; ++k) {
    float mu  = stats[k] * invB;
    float var = stats[8 + k] * invB - mu * mu;
    float sc  = bn_g[k] * rsqrtf(var + 1e-5f);
    scale[k] = sc;
    shift[k] = bn_b[k] - mu * sc;
  }

  // ---- weight(+bias) fragments in A layout, resident in VGPRs ----
  v16h Afm = make_wA(w_fm, b_fm,  8, 16, lane);
  v16h Ac1 = make_wA(w_c1, b_c1, 16, 16, lane);
  v16h Ap1 = make_wA(w_p1, b_p1, 16, 12, lane);
  v16h Ac2 = make_wA(w_c2, b_c2, 12,  8, lane);
  v16h Ap2 = make_wA(w_p2, b_p2,  8,  4, lane);
  v16h Ac3 = make_wA(w_c3, b_c3,  4,  4, lane);
  v16h Ar  = make_wA(w_r,  b_r,   4,  4, lane);
  v16h Ah  = make_wA(w_h,  b_h,   4,  1, lane);

  // persistent all-zero accumulator input (lowers to inline SRC2 = 0)
  v8f zc;
#pragma unroll
  for (int r = 0; r < 8; ++r) zc[r] = 0.0f;

  const int gwave = blockIdx.x * WAVES_PER_BLOCK + wave;

  for (int t = 0; t < TPW; ++t) {
    const int row0 = (gwave * TPW + t) * 32;   // wave-uniform; 2 sub-tiles = 32 rows
    if (row0 + 32 > nrows) break;

    // ---- input B fragments: X^T (K = feature 0..7), 1.0 at K==8 (bias) ----
    v16h b0, b1;
#pragma unroll
    for (int i = 0; i < 16; ++i) { b0[i] = (_Float16)0.0f; b1[i] = (_Float16)0.0f; }
    if (hb == 0) {
      const float4* p0 = (const float4*)(x + (size_t)(row0 + n) * 8);
      const float4* p1 = (const float4*)(x + (size_t)(row0 + 16 + n) * 8);
      float4 lo0 = p0[0], hi0 = p0[1];
      float4 lo1 = p1[0], hi1 = p1[1];
      float v0[8] = {lo0.x, lo0.y, lo0.z, lo0.w, hi0.x, hi0.y, hi0.z, hi0.w};
      float v1[8] = {lo1.x, lo1.y, lo1.z, lo1.w, hi1.x, hi1.y, hi1.z, hi1.w};
#pragma unroll
      for (int k = 0; k < 8; ++k) {
        b0[k] = (_Float16)(v0[k] * scale[k] + shift[k]);
        b1[k] = (_Float16)(v1[k] * scale[k] + shift[k]);
      }
      b0[8] = (_Float16)1.0f;
      b1[8] = (_Float16)1.0f;
    }

    v8f c0, c1, act0, act1;

#define GEMM(AW)                                                                      \
    c0 = __builtin_amdgcn_wmma_f32_16x16x32_f16(false, AW, false, b0, (short)0, zc, false, false); \
    c1 = __builtin_amdgcn_wmma_f32_16x16x32_f16(false, AW, false, b1, (short)0, zc, false, false);

#define TANH_N(NT)                                                                    \
    _Pragma("unroll") for (int r = 0; r < (NT); ++r) {                                \
      act0[r] = fast_tanh(c0[r]); act1[r] = fast_tanh(c1[r]);                         \
    }

    GEMM(Afm); TANH_N(8);                          // fm: 8 -> 16
    b0 = next_b<16>(act0, hb); b1 = next_b<16>(act1, hb);
    GEMM(Ac1); TANH_N(8);                          // c1: 16 -> 16
    b0 = next_b<16>(act0, hb); b1 = next_b<16>(act1, hb);
    GEMM(Ap1); TANH_N(8);                          // p1: 16 -> 12
    b0 = next_b<12>(act0, hb); b1 = next_b<12>(act1, hb);
    GEMM(Ac2); TANH_N(8);                          // c2: 12 -> 8
    b0 = next_b<8>(act0, hb);  b1 = next_b<8>(act1, hb);
    GEMM(Ap2); TANH_N(4);                          // p2: 8 -> 4
    b0 = next_b<4>(act0, hb);  b1 = next_b<4>(act1, hb);

    GEMM(Ac3);                                     // c3: 4 -> 4 (keep h5)
    v8f h50, h51;
#pragma unroll
    for (int r = 0; r < 4; ++r) { h50[r] = fast_tanh(c0[r]); h51[r] = fast_tanh(c1[r]); }
    b0 = next_b<4>(h50, hb);   b1 = next_b<4>(h51, hb);

    GEMM(Ar);                                      // residual: h6 = h5 + relu(.)
    v8f h60, h61;
#pragma unroll
    for (int r = 0; r < 4; ++r) {
      h60[r] = h50[r] + fmaxf(c0[r], 0.0f);
      h61[r] = h51[r] + fmaxf(c1[r], 0.0f);
    }
    b0 = next_b<4>(h60, hb);   b1 = next_b<4>(h61, hb);

    GEMM(Ah);                                      // head: 4 -> 1, sigmoid
    if (hb == 0) {                                 // M=0 row lives in lanes 0-15, reg 0
      out[row0 + n]      = fast_sigmoid(c0[0]);
      out[row0 + 16 + n] = fast_sigmoid(c1[0]);
    }
#undef GEMM
#undef TANH_N
  }
}

extern "C" void kernel_launch(void* const* d_in, const int* in_sizes, int n_in,
                              void* d_out, int out_size, void* d_ws, size_t ws_size,
                              hipStream_t stream) {
  const float* x    = (const float*)d_in[0];
  const float* bn_g = (const float*)d_in[1];
  const float* bn_b = (const float*)d_in[2];
  const float* w_fm = (const float*)d_in[3];
  const float* b_fm = (const float*)d_in[4];
  const float* w_c1 = (const float*)d_in[5];
  const float* b_c1 = (const float*)d_in[6];
  const float* w_p1 = (const float*)d_in[7];
  const float* b_p1 = (const float*)d_in[8];
  const float* w_c2 = (const float*)d_in[9];
  const float* b_c2 = (const float*)d_in[10];
  const float* w_p2 = (const float*)d_in[11];
  const float* b_p2 = (const float*)d_in[12];
  const float* w_c3 = (const float*)d_in[13];
  const float* b_c3 = (const float*)d_in[14];
  const float* w_r  = (const float*)d_in[15];
  const float* b_r  = (const float*)d_in[16];
  const float* w_h  = (const float*)d_in[17];
  const float* b_h  = (const float*)d_in[18];

  float* ws  = (float*)d_ws;
  float* out = (float*)d_out;
  const int nrows = in_sizes[0] / 8;

  zero_ws<<<1, 32, 0, stream>>>(ws);
  bn_stats<<<1024, 256, 0, stream>>>(x, ws, nrows);

  const int rowsPerBlock = WAVES_PER_BLOCK * TPW * 32;   // 2048
  const int blocks = (nrows + rowsPerBlock - 1) / rowsPerBlock;  // 512 for B=1M
  qcnn_fused<<<blocks, 256, 0, stream>>>(
      x, bn_g, bn_b, w_fm, b_fm, w_c1, b_c1, w_p1, b_p1, w_c2, b_c2,
      w_p2, b_p2, w_c3, b_c3, w_r, b_r, w_h, b_h, ws, out, nrows);
}
